// VectorMerge_41180146434605
// MI455X (gfx1250) — matrix operand order repaired and checked by reading, not compiled, not run on hardware
//
#include <hip/hip_runtime.h>

// ---------------------------------------------------------------------------
// Types
// ---------------------------------------------------------------------------
typedef __attribute__((ext_vector_type(16))) __bf16        v16bf;
typedef __attribute__((ext_vector_type(8)))  float         v8f;
typedef __attribute__((ext_vector_type(4)))  unsigned int  v4u;

union Frag {
    v16bf v;
    v4u   q[2];
};

__device__ __forceinline__ unsigned short f2bf(float f) {
    union { float f; unsigned int u; } c;
    c.f = f;
    unsigned int r = c.u + 0x7FFFu + ((c.u >> 16) & 1u);  // round-to-nearest-even
    return (unsigned short)(r >> 16);
}

// ---------------------------------------------------------------------------
// CDNA5 async-to-LDS copy (ASYNCcnt-tracked, bypasses VGPR staging)
// ---------------------------------------------------------------------------
__device__ __forceinline__ void async_copy_b128(const void* gptr, void* lptr) {
    // LDS aperture keeps the wave-relative LDS offset in addr[31:0]
    unsigned           lds = (unsigned)(unsigned long long)lptr;
    unsigned long long g   = (unsigned long long)gptr;
    asm volatile("global_load_async_to_lds_b128 %0, %1, off"
                 :: "v"(lds), "v"(g)
                 : "memory");
}

template <int N>
__device__ __forceinline__ void wait_asynccnt() {
    asm volatile("s_wait_asynccnt %0" :: "n"(N) : "memory");
}

// ---------------------------------------------------------------------------
// Stage 1: LayerNorm -> leaky_relu -> bf16   (one block per row)
// ---------------------------------------------------------------------------
template <int VPT>
__global__ __launch_bounds__(256)
void ln_act_kernel(const float* __restrict__ x,
                   const float* __restrict__ sc,
                   const float* __restrict__ bi,
                   unsigned short* __restrict__ f, int D)
{
    __shared__ float red[256];
    const int row = blockIdx.x;
    const int t   = threadIdx.x;
    const float* xr = x + (size_t)row * D;

    float xv[VPT];
    float s = 0.f;
#pragma unroll
    for (int i = 0; i < VPT; ++i) { xv[i] = xr[t + i * 256]; s += xv[i]; }

    red[t] = s; __syncthreads();
    for (int off = 128; off > 0; off >>= 1) {
        if (t < off) red[t] += red[t + off];
        __syncthreads();
    }
    const float mean = red[0] * (1.0f / D);
    __syncthreads();

    float vs = 0.f;
#pragma unroll
    for (int i = 0; i < VPT; ++i) { float d = xv[i] - mean; vs += d * d; }
    red[t] = vs; __syncthreads();
    for (int off = 128; off > 0; off >>= 1) {
        if (t < off) red[t] += red[t + off];
        __syncthreads();
    }
    const float rstd = rsqrtf(red[0] * (1.0f / D) + 1e-6f);

#pragma unroll
    for (int i = 0; i < VPT; ++i) {
        const int idx = t + i * 256;
        float y = (xv[i] - mean) * rstd * sc[idx] + bi[idx];
        y = (y > 0.f) ? y : 0.01f * y;           // leaky_relu, slope 0.01
        f[(size_t)row * D + idx] = f2bf(y);
    }
}

// ---------------------------------------------------------------------------
// Stage 2: W[K][N] fp32  ->  Wt[N][K] bf16   (32x32 LDS transpose tiles)
// ---------------------------------------------------------------------------
__global__ __launch_bounds__(256)
void wt_kernel(const float* __restrict__ W,
               unsigned short* __restrict__ Wt, int K, int N)
{
    __shared__ float tile[32][33];
    const int k0 = blockIdx.y * 32;
    const int n0 = blockIdx.x * 32;
    const int tx = threadIdx.x & 31;
    const int ty = threadIdx.x >> 5;      // 0..7

#pragma unroll
    for (int i = 0; i < 32; i += 8)
        tile[ty + i][tx] = W[(size_t)(k0 + ty + i) * N + n0 + tx];
    __syncthreads();
#pragma unroll
    for (int i = 0; i < 32; i += 8)
        Wt[(size_t)(n0 + ty + i) * K + k0 + tx] = f2bf(tile[tx][ty + i]);
}

// ---------------------------------------------------------------------------
// Stage 3: fused dual-accumulator bf16 WMMA GEMM + gated epilogue
//   out[b,n] = sigmoid(f1@Wg1 + f2@Wg2 + bg) * (f1@Wo1 + f2@Wo2 + bo)
// Block tile 128x128, 8 waves (2x4), wave tile 64x32, K-step 32.
// Double-buffered LDS, filled by async-to-LDS DMA copies (ASYNCcnt).
// ---------------------------------------------------------------------------
#define BM     128
#define BN     128
#define KS     32
#define LDT    40        // padded LDS row stride (bf16) -> conflict-free b128
#define STAGES 48        // 32 K-slices from f1 (D1=1024) + 16 from f2 (D2=512)

__global__ __launch_bounds__(256)
void fused_gemm_kernel(const unsigned short* __restrict__ f1,
                       const unsigned short* __restrict__ f2,
                       const unsigned short* __restrict__ wo1,
                       const unsigned short* __restrict__ wg1,
                       const unsigned short* __restrict__ wo2,
                       const unsigned short* __restrict__ wg2,
                       const float* __restrict__ b_o1,
                       const float* __restrict__ b_o2,
                       const float* __restrict__ b_g1,
                       const float* __restrict__ b_g2,
                       float* __restrict__ out)
{
    __shared__ __align__(16) unsigned short As[2][BM * LDT];        // 2 x 10 KB
    __shared__ __align__(16) unsigned short Bs[2][2 * BN * LDT];    // 2 x 20 KB

    const int tid  = threadIdx.x;
    const int lane = tid & 31;
    const int wave = tid >> 5;
    const int wm0  = (wave >> 2) * 64;   // 2 wave-rows
    const int wn0  = (wave & 3) * 32;    // 4 wave-cols
    const int half = lane >> 4;
    const int r16  = lane & 15;

    const int bm = blockIdx.y * BM;
    const int bn = blockIdx.x * BN;

    // per-thread DMA coordinates: each thread copies 2 A rows + 2x2 B rows,
    // one 16B (8 x bf16) chunk each -> 6 async b128 copies per stage
    const int cpy_row = tid >> 2;          // 0..63 (+64 on second unit)
    const int cpy_kc  = (tid & 3) << 3;    // 0,8,16,24

    auto issue_stage = [&](int s, int buf) {
        const int ph = (s >= 32) ? 1 : 0;
        const unsigned short* fa = ph ? f2  : f1;
        const unsigned short* wo = ph ? wo2 : wo1;
        const unsigned short* wg = ph ? wg2 : wg1;
        const int Kd = ph ? 512 : 1024;
        const int k0 = (ph ? (s - 32) : s) * KS;
#pragma unroll
        for (int i = 0; i < 2; ++i) {
            const int row = cpy_row + i * 64;
            async_copy_b128(&fa[(size_t)(bm + row) * Kd + k0 + cpy_kc],
                            &As[buf][row * LDT + cpy_kc]);
        }
#pragma unroll
        for (int i = 0; i < 2; ++i) {
            const int n = cpy_row + i * 64;
            async_copy_b128(&wo[(size_t)(bn + n) * Kd + k0 + cpy_kc],
                            &Bs[buf][n * LDT + cpy_kc]);
            async_copy_b128(&wg[(size_t)(bn + n) * Kd + k0 + cpy_kc],
                            &Bs[buf][BN * LDT + n * LDT + cpy_kc]);
        }
    };

    v8f acc_o[4][2], acc_g[4][2];
#pragma unroll
    for (int i = 0; i < 4; ++i)
#pragma unroll
        for (int j = 0; j < 2; ++j)
#pragma unroll
            for (int e = 0; e < 8; ++e) { acc_o[i][j][e] = 0.f; acc_g[i][j][e] = 0.f; }

    issue_stage(0, 0);

#pragma unroll 1
    for (int s = 0; s < STAGES; ++s) {
        const int buf = s & 1;
        if (s + 1 < STAGES) {
            issue_stage(s + 1, buf ^ 1);   // prefetch runs under this stage's WMMAs
            wait_asynccnt<6>();            // stage s's 6 copies retired (in-order)
        } else {
            wait_asynccnt<0>();
        }
        __syncthreads();                    // all waves' stage-s data visible

        // ---- fragments
        // A 16x32 bf16: lane<16 -> K0..15 contiguous, lane>=16 -> K16..31
        Frag a[4];
#pragma unroll
        for (int mt = 0; mt < 4; ++mt) {
            const int base = (wm0 + mt * 16 + r16) * LDT;
            a[mt].q[0] = *(const v4u*)(&As[buf][base + half * 16]);
            a[mt].q[1] = *(const v4u*)(&As[buf][base + half * 16 + 8]);
        }
        // B 32x16 bf16: lane<16 -> K0..15 contiguous, lane>=16 -> K16..31
        Frag bo[2], bg[2];
#pragma unroll
        for (int nt = 0; nt < 2; ++nt) {
            const int base = (wn0 + nt * 16 + r16) * LDT + half * 16;
            bo[nt].q[0] = *(const v4u*)(&Bs[buf][base]);
            bo[nt].q[1] = *(const v4u*)(&Bs[buf][base + 8]);
            bg[nt].q[0] = *(const v4u*)(&Bs[buf][BN * LDT + base]);
            bg[nt].q[1] = *(const v4u*)(&Bs[buf][BN * LDT + base + 8]);
        }

        // ---- 16 WMMAs per stage per wave (A reused 4x)
#pragma unroll
        for (int mt = 0; mt < 4; ++mt)
#pragma unroll
            for (int nt = 0; nt < 2; ++nt) {
                acc_o[mt][nt] = __builtin_amdgcn_wmma_f32_16x16x32_bf16(
                    false, a[mt].v, false, bo[nt].v,
                    (short)0, acc_o[mt][nt], false, false);
                acc_g[mt][nt] = __builtin_amdgcn_wmma_f32_16x16x32_bf16(
                    false, a[mt].v, false, bg[nt].v,
                    (short)0, acc_g[mt][nt], false, false);
            }

        __syncthreads();                    // release buf for the next prefetch
    }

    // ---- epilogue: C/D layout -> VGPR e holds M = e + 8*half, N = lane&15
#pragma unroll
    for (int nt = 0; nt < 2; ++nt) {
        const int col = bn + wn0 + nt * 16 + r16;
        const float bsum_o = b_o1[col] + b_o2[col];
        const float bsum_g = b_g1[col] + b_g2[col];
#pragma unroll
        for (int mt = 0; mt < 4; ++mt) {
#pragma unroll
            for (int e = 0; e < 8; ++e) {
                const int row = bm + wm0 + mt * 16 + half * 8 + e;
                const float o = acc_o[mt][nt][e] + bsum_o;
                const float g = acc_g[mt][nt][e] + bsum_g;
                const float sg = 1.0f / (1.0f + __expf(-g));
                out[(size_t)row * 1024 + col] = sg * o;
            }
        }
    }
}

// ---------------------------------------------------------------------------
// Launch
// ---------------------------------------------------------------------------
extern "C" void kernel_launch(void* const* d_in, const int* in_sizes, int n_in,
                              void* d_out, int out_size, void* d_ws, size_t ws_size,
                              hipStream_t stream)
{
    (void)in_sizes; (void)n_in; (void)out_size; (void)ws_size;

    constexpr int B = 32768, D1 = 1024, D2 = 512, OUT = 1024;

    const float* x1     = (const float*)d_in[0];
    const float* x2     = (const float*)d_in[1];
    const float* s1     = (const float*)d_in[2];
    const float* bb1    = (const float*)d_in[3];
    const float* s2     = (const float*)d_in[4];
    const float* bb2    = (const float*)d_in[5];
    const float* W_out1 = (const float*)d_in[6];
    const float* b_out1 = (const float*)d_in[7];
    const float* W_out2 = (const float*)d_in[8];
    const float* b_out2 = (const float*)d_in[9];
    const float* W_g1   = (const float*)d_in[10];
    const float* b_g1   = (const float*)d_in[11];
    const float* W_g2   = (const float*)d_in[12];
    const float* b_g2   = (const float*)d_in[13];
    float* out = (float*)d_out;

    // workspace layout (bf16 = 2B each), ~102 MB total
    char* ws = (char*)d_ws;
    size_t off = 0;
    unsigned short* f1  = (unsigned short*)(ws + off); off += (size_t)B * D1 * 2;   // 64 MB
    unsigned short* f2  = (unsigned short*)(ws + off); off += (size_t)B * D2 * 2;   // 32 MB
    unsigned short* wo1 = (unsigned short*)(ws + off); off += (size_t)OUT * D1 * 2; //  2 MB
    unsigned short* wg1 = (unsigned short*)(ws + off); off += (size_t)OUT * D1 * 2;
    unsigned short* wo2 = (unsigned short*)(ws + off); off += (size_t)OUT * D2 * 2; //  1 MB
    unsigned short* wg2 = (unsigned short*)(ws + off); off += (size_t)OUT * D2 * 2;

    // Stage 1: LN + leaky_relu -> bf16
    ln_act_kernel<4><<<B, 256, 0, stream>>>(x1, s1, bb1, f1, D1);
    ln_act_kernel<2><<<B, 256, 0, stream>>>(x2, s2, bb2, f2, D2);

    // Stage 2: weight transpose + bf16 convert  (Wt[N][K])
    wt_kernel<<<dim3(OUT / 32, D1 / 32), 256, 0, stream>>>(W_out1, wo1, D1, OUT);
    wt_kernel<<<dim3(OUT / 32, D1 / 32), 256, 0, stream>>>(W_g1,   wg1, D1, OUT);
    wt_kernel<<<dim3(OUT / 32, D2 / 32), 256, 0, stream>>>(W_out2, wo2, D2, OUT);
    wt_kernel<<<dim3(OUT / 32, D2 / 32), 256, 0, stream>>>(W_g2,   wg2, D2, OUT);

    // Stage 3: fused dual GEMM + gate epilogue (async double-buffered WMMA)
    fused_gemm_kernel<<<dim3(OUT / BN, B / BM), 256, 0, stream>>>(
        f1, f2, wo1, wg1, wo2, wg2, b_out1, b_out2, b_g1, b_g2, out);
}
